// BertBiLayerGCN_61074434949547
// MI455X (gfx1250) — matrix operand order repaired and checked by reading, not compiled, not run on hardware
//
#include <hip/hip_runtime.h>
#include <math.h>

// ---------------------------------------------------------------------------
// Types for CDNA5 WMMA
// ---------------------------------------------------------------------------
typedef __attribute__((ext_vector_type(16))) __bf16 v16bf;
typedef __attribute__((ext_vector_type(8)))  float  v8f;

typedef unsigned short u16;
typedef unsigned int   u32;
typedef unsigned long long u64;
typedef __attribute__((address_space(3))) u16 lds_u16;

#define NBATCH 2048
#define SEQ    64
#define HID    768
#define NHEADS 12
#define DHEAD  64
#define FFD    3072
#define MTOK   (NBATCH * SEQ)      // 131072
#define EG     32768
#define EC     65536
#define G1DIM  128
#define G2DIM  64

// ---------------------------------------------------------------------------
// bf16 helpers (round-to-nearest-even)
// ---------------------------------------------------------------------------
__device__ __forceinline__ u16 f2bf(float f) {
    u32 u = __float_as_uint(f);
    u32 r = u + 0x7FFFu + ((u >> 16) & 1u);
    return (u16)(r >> 16);
}
__device__ __forceinline__ float bf2f(u16 h) {
    return __uint_as_float(((u32)h) << 16);
}
__device__ __forceinline__ float gelu_tanh(float x) {
    float x3 = x * x * x;
    return 0.5f * x * (1.0f + tanhf(0.7978845608028654f * (x + 0.044715f * x3)));
}

// ---------------------------------------------------------------------------
// WMMA fragment loaders (bf16, 16x16x32), per CDNA5 ISA §7.12.2
// A 16x32:  lanes 0-15 -> M=lane, K-groups {0..7 | 16..23};
//           lanes 16-31 -> same M, K-groups {8..15 | 24..31}
// B 32x16:  lanes 0-15 -> N=lane, K-groups {0..7 | 16..23} (pairs across 8 VGPRs);
//           lanes 16-31 -> same N, K-groups {8..15 | 24..31}
// ---------------------------------------------------------------------------
__device__ __forceinline__ v16bf load_frag_a(const u16* base, int stride, int lane) {
    const int m  = lane & 15;
    const int kb = (lane >> 4) << 3;     // 0 or 8
    const u16* row = base + m * stride;
    union { v16bf v; u32 u[8]; } f;
#pragma unroll
    for (int i = 0; i < 4; i++) {
        f.u[i]     = *(const u32*)(row + kb + 2 * i);
        f.u[4 + i] = *(const u32*)(row + kb + 16 + 2 * i);
    }
    return f.v;
}
__device__ __forceinline__ v16bf load_frag_b(const u16* base, int stride, int lane) {
    const int n  = lane & 15;
    const int kb = (lane >> 4) << 3;     // 0 or 8
    const u16* row = base + n * stride;
    union { v16bf v; u32 u[8]; } f;
#pragma unroll
    for (int i = 0; i < 4; i++) {
        f.u[i]     = *(const u32*)(row + kb + 2 * i);
        f.u[4 + i] = *(const u32*)(row + kb + 16 + 2 * i);
    }
    return f.v;
}

// ---------------------------------------------------------------------------
// CDNA5 async global->LDS staging (bypasses VGPRs, tracked by ASYNCcnt).
// Each call issues 4 x global_load_async_to_lds_b128 per lane:
// 256 threads x 4 x 16B = one full 128x64 bf16 tile.
// ---------------------------------------------------------------------------
#define TK   64
#define APAD 8
#define LDST (TK + APAD)                 // 72 u16 = 144 B row stride (9*16B)

#define WAIT_ASYNC(n) asm volatile("s_wait_asynccnt " #n ::: "memory")

__device__ __forceinline__ void stage_tile_async(const u16* gbase, int k0, size_t ld,
                                                 u16 (*buf)[LDST], int tid)
{
    const int rrow = tid >> 3;           // 0..31
    const int seg  = (tid & 7) * 8;      // 0..56 u16 (16B granules)
#pragma unroll
    for (int it = 0; it < 4; it++) {
        const int row = rrow + 32 * it;
        u64 ga = (u64)(uintptr_t)(gbase + (size_t)row * ld + k0 + seg);
        u32 la = (u32)(uintptr_t)(lds_u16*)(&buf[row][seg]);
        asm volatile("global_load_async_to_lds_b128 %0, %1, off"
                     :: "v"(la), "v"(ga) : "memory");
    }
}

// ---------------------------------------------------------------------------
// Generic tiled WMMA GEMM:  C[M,N] = act( A[M,K] @ Bt[N,K]^T + bias + resid )
// A row-major bf16 (row stride lda), Bt row-major bf16 [N,K] (row stride ldb),
// bias fp32 [N] (nullable), resid bf16 same layout as C (nullable),
// act: 0 none, 1 gelu(tanh), 2 tanh.
// Grid: (M/128, N/128), block 256 threads (8 wave32).
// Double-buffered LDS, async-to-LDS staging, 16 WMMAs per K-tile of 64.
// ---------------------------------------------------------------------------
__global__ __launch_bounds__(256) void gemm_bf16_wmma(
    const u16* __restrict__ A, int lda,
    const u16* __restrict__ Bt, int ldb,
    const float* __restrict__ bias,
    const u16* __restrict__ resid, int ldr,
    u16* __restrict__ C, int ldc,
    int K, int act)
{
    __shared__ u16 As[2][128][LDST];
    __shared__ u16 Bs[2][128][LDST];

    const int tid  = threadIdx.x;
    const int lane = tid & 31;
    const int wave = tid >> 5;
    const size_t bm = (size_t)blockIdx.x * 128;
    const size_t bn = (size_t)blockIdx.y * 128;
    const int wm = (wave >> 1) * 32;     // 4 waves over M
    const int wn = (wave & 1) * 64;      // 2 waves over N

    v8f acc[2][4] = {};

    const u16* gA = A  + bm * (size_t)lda;
    const u16* gB = Bt + bn * (size_t)ldb;

    // prime buffer 0
    stage_tile_async(gA, 0, (size_t)lda, As[0], tid);
    stage_tile_async(gB, 0, (size_t)ldb, Bs[0], tid);

    int cur = 0;
    for (int k0 = 0; k0 < K; k0 += TK) {
        __syncthreads();                 // all waves done reading buf[1-cur]
        if (k0 + TK < K) {
            stage_tile_async(gA, k0 + TK, (size_t)lda, As[1 - cur], tid);
            stage_tile_async(gB, k0 + TK, (size_t)ldb, Bs[1 - cur], tid);
            WAIT_ASYNC(0x8);             // 8 newest = prefetch; current tile landed
        } else {
            WAIT_ASYNC(0x0);
        }
        __syncthreads();                 // current tile visible to all waves

#pragma unroll
        for (int kc = 0; kc < 2; kc++) {
            v16bf af[2], bfr[4];
#pragma unroll
            for (int i = 0; i < 2; i++)
                af[i] = load_frag_a(&As[cur][wm + 16 * i][kc * 32], LDST, lane);
#pragma unroll
            for (int j = 0; j < 4; j++)
                bfr[j] = load_frag_b(&Bs[cur][wn + 16 * j][kc * 32], LDST, lane);
#pragma unroll
            for (int i = 0; i < 2; i++)
#pragma unroll
                for (int j = 0; j < 4; j++)
                    acc[i][j] = __builtin_amdgcn_wmma_f32_16x16x32_bf16(
                        false, af[i], false, bfr[j], (short)0, acc[i][j], false, false);
        }
        cur ^= 1;
    }

    const int nlo = lane & 15;
    const int mhi = (lane >> 4) * 8;
#pragma unroll
    for (int i = 0; i < 2; i++) {
#pragma unroll
        for (int j = 0; j < 4; j++) {
            size_t col = bn + wn + 16 * j + nlo;
            float bv = bias ? bias[col] : 0.0f;
#pragma unroll
            for (int rr = 0; rr < 8; rr++) {
                size_t row = bm + wm + 16 * i + rr + mhi;
                float v = acc[i][j][rr] + bv;
                if (resid) v += bf2f(resid[row * (size_t)ldr + col]);
                if (act == 1)      v = gelu_tanh(v);
                else if (act == 2) v = tanhf(v);
                C[row * (size_t)ldc + col] = f2bf(v);
            }
        }
    }
}

// ---------------------------------------------------------------------------
// Attention: one workgroup (4 wave32) per (batch, head). S=64, DH=64.
//   scores = Q K^T / 8 + mask-bias ; softmax ; ctx = P V
// ---------------------------------------------------------------------------
__global__ __launch_bounds__(128) void attention_kernel(
    const u16* __restrict__ Q, const u16* __restrict__ Km,
    const u16* __restrict__ Vm, const int* __restrict__ mask,
    u16* __restrict__ Ctx)
{
    const int bh = blockIdx.x;
    const int b  = bh / NHEADS;
    const int h  = bh - b * NHEADS;
    const int tid  = threadIdx.x;
    const int lane = tid & 31;
    const int wave = tid >> 5;

    __shared__ u16   Ks[64][72];          // K rows (s, d)
    __shared__ u16   Vt[64][72];          // V transposed (d, s)
    __shared__ float Ps[4][16][64];       // per-wave score strips (fp32)
    __shared__ u16   Pb[4][16][72];       // per-wave softmaxed strips (bf16)

    // ---- stage K (row-major, vectorized) ----
    {
        const int s  = tid >> 1;
        const int sg = (tid & 1) * 32;
        const u16* src = Km + ((size_t)(b * SEQ + s)) * HID + h * DHEAD + sg;
#pragma unroll
        for (int q4 = 0; q4 < 4; q4++)
            *(uint4*)&Ks[s][sg + q4 * 8] = *(const uint4*)(src + q4 * 8);
    }
    // ---- stage V transposed (scalar) ----
    for (int idx = 0; idx < 32; idx++) {
        int lin = tid * 32 + idx;        // 0..4095
        int s = lin >> 6, d = lin & 63;
        Vt[d][s] = Vm[((size_t)(b * SEQ + s)) * HID + h * DHEAD + d];
    }
    __syncthreads();

    const int s0 = wave * 16;            // this wave's 16 query rows

    // ---- scores = Q K^T ----
    v8f sacc[4] = {};
#pragma unroll
    for (int ch = 0; ch < 2; ch++) {
        const u16* qrow = Q + ((size_t)(b * SEQ + s0)) * HID + h * DHEAD + ch * 32;
        v16bf qa = load_frag_a(qrow, HID, lane);
#pragma unroll
        for (int j = 0; j < 4; j++) {
            v16bf kb = load_frag_b(&Ks[16 * j][ch * 32], 72, lane);
            sacc[j] = __builtin_amdgcn_wmma_f32_16x16x32_bf16(
                false, qa, false, kb, (short)0, sacc[j], false, false);
        }
    }
    // scale + mask bias -> LDS strip
    {
        const int nlo = lane & 15;
        const int mhi = (lane >> 4) * 8;
#pragma unroll
        for (int j = 0; j < 4; j++) {
            int col = 16 * j + nlo;
            float bv = (1.0f - (float)mask[b * SEQ + col]) * -10000.0f;
#pragma unroll
            for (int rr = 0; rr < 8; rr++)
                Ps[wave][rr + mhi][col] = sacc[j][rr] * 0.125f + bv;
        }
    }
    __syncthreads();

    // ---- per-row softmax (lanes 0-15 of each wave own one row each) ----
    if (lane < 16) {
        float mx = -3.4e38f;
        for (int c = 0; c < 64; c++) mx = fmaxf(mx, Ps[wave][lane][c]);
        float sum = 0.0f;
        for (int c = 0; c < 64; c++) sum += __expf(Ps[wave][lane][c] - mx);
        float inv = 1.0f / sum;
        for (int c = 0; c < 64; c++)
            Pb[wave][lane][c] = f2bf(__expf(Ps[wave][lane][c] - mx) * inv);
    }
    __syncthreads();

    // ---- ctx = P V ----
    v8f cacc[4] = {};
#pragma unroll
    for (int ch = 0; ch < 2; ch++) {
        v16bf pa = load_frag_a(&Pb[wave][0][ch * 32], 72, lane);
#pragma unroll
        for (int j = 0; j < 4; j++) {
            v16bf vb = load_frag_b(&Vt[16 * j][ch * 32], 72, lane);
            cacc[j] = __builtin_amdgcn_wmma_f32_16x16x32_bf16(
                false, pa, false, vb, (short)0, cacc[j], false, false);
        }
    }
    {
        const int nlo = lane & 15;
        const int mhi = (lane >> 4) * 8;
#pragma unroll
        for (int j = 0; j < 4; j++) {
            int d = 16 * j + nlo;
#pragma unroll
            for (int rr = 0; rr < 8; rr++) {
                int srow = s0 + rr + mhi;
                Ctx[((size_t)(b * SEQ + srow)) * HID + h * DHEAD + d] = f2bf(cacc[j][rr]);
            }
        }
    }
}

// ---------------------------------------------------------------------------
// Embedding gather + LayerNorm (block = one token row, 256 threads x 3 cols)
// ---------------------------------------------------------------------------
__global__ __launch_bounds__(256) void embed_ln_kernel(
    const int* __restrict__ ids, const int* __restrict__ seg,
    const float* __restrict__ word, const float* __restrict__ pos,
    const float* __restrict__ typ, const float* __restrict__ g,
    const float* __restrict__ be, u16* __restrict__ H)
{
    const int m = blockIdx.x;
    const int s = m & (SEQ - 1);
    const int id = ids[m];
    const int tg = seg[m];
    const int tid = threadIdx.x;
    __shared__ float red[256];

    float v[3];
    float sum = 0.0f;
#pragma unroll
    for (int i = 0; i < 3; i++) {
        int c = tid + 256 * i;
        v[i] = word[(size_t)id * HID + c] + pos[(size_t)s * HID + c] + typ[(size_t)tg * HID + c];
        sum += v[i];
    }
    red[tid] = sum; __syncthreads();
    for (int st = 128; st > 0; st >>= 1) {
        if (tid < st) red[tid] += red[tid + st];
        __syncthreads();
    }
    float mean = red[0] * (1.0f / HID);
    __syncthreads();
    float vs = 0.0f;
#pragma unroll
    for (int i = 0; i < 3; i++) { float d = v[i] - mean; vs += d * d; }
    red[tid] = vs; __syncthreads();
    for (int st = 128; st > 0; st >>= 1) {
        if (tid < st) red[tid] += red[tid + st];
        __syncthreads();
    }
    float inv = rsqrtf(red[0] * (1.0f / HID) + 1e-12f);
#pragma unroll
    for (int i = 0; i < 3; i++) {
        int c = tid + 256 * i;
        H[(size_t)m * HID + c] = f2bf((v[i] - mean) * inv * g[c] + be[c]);
    }
}

// LayerNorm over bf16 rows of width 768
__global__ __launch_bounds__(256) void ln_kernel(
    const u16* __restrict__ X, const float* __restrict__ g,
    const float* __restrict__ be, u16* __restrict__ Y)
{
    const int m = blockIdx.x;
    const int tid = threadIdx.x;
    __shared__ float red[256];
    float v[3];
    float sum = 0.0f;
#pragma unroll
    for (int i = 0; i < 3; i++) {
        int c = tid + 256 * i;
        v[i] = bf2f(X[(size_t)m * HID + c]);
        sum += v[i];
    }
    red[tid] = sum; __syncthreads();
    for (int st = 128; st > 0; st >>= 1) { if (tid < st) red[tid] += red[tid + st]; __syncthreads(); }
    float mean = red[0] * (1.0f / HID);
    __syncthreads();
    float vs = 0.0f;
#pragma unroll
    for (int i = 0; i < 3; i++) { float d = v[i] - mean; vs += d * d; }
    red[tid] = vs; __syncthreads();
    for (int st = 128; st > 0; st >>= 1) { if (tid < st) red[tid] += red[tid + st]; __syncthreads(); }
    float inv = rsqrtf(red[0] * (1.0f / HID) + 1e-12f);
#pragma unroll
    for (int i = 0; i < 3; i++) {
        int c = tid + 256 * i;
        Y[(size_t)m * HID + c] = f2bf((v[i] - mean) * inv * g[c] + be[c]);
    }
}

// fp32 [K,N] -> bf16 transposed [N,K]
__global__ void cvt_transpose_kernel(const float* __restrict__ W, u16* __restrict__ Wt,
                                     int K, int Nn)
{
    int i = blockIdx.x * 256 + threadIdx.x;
    if (i >= K * Nn) return;
    int k = i / Nn, n = i - k * Nn;
    Wt[(size_t)n * K + k] = f2bf(W[i]);
}

// ---------------------------------------------------------------------------
// GCN kernels (tiny FLOPs; plain VALU)
// ---------------------------------------------------------------------------
__global__ void gemm_bf16f32_plain(const u16* __restrict__ A, const float* __restrict__ W,
                                   float* __restrict__ out, int Mr, int K, int Nn)
{
    int t = blockIdx.x * 256 + threadIdx.x;
    if (t >= Mr * Nn) return;
    int m = t / Nn, n = t - m * Nn;
    float s = 0.0f;
    for (int k = 0; k < K; k++) s += bf2f(A[(size_t)m * K + k]) * W[(size_t)k * Nn + n];
    out[t] = s;
}
__global__ void gemm_f32_plain(const float* __restrict__ A, const float* __restrict__ W,
                               float* __restrict__ out, int Mr, int K, int Nn)
{
    int t = blockIdx.x * 256 + threadIdx.x;
    if (t >= Mr * Nn) return;
    int m = t / Nn, n = t - m * Nn;
    float s = 0.0f;
    for (int k = 0; k < K; k++) s += A[(size_t)m * K + k] * W[(size_t)k * Nn + n];
    out[t] = s;
}
__global__ void deg_kernel(const int* __restrict__ ei, float* __restrict__ deg)
{
    int e = blockIdx.x * 256 + threadIdx.x;
    if (e >= EG + NBATCH) return;
    int c = (e < EG) ? ei[EG + e] : (e - EG);
    atomicAdd(&deg[c], 1.0f);
}
__global__ void scatter_kernel(const int* __restrict__ ei, const float* __restrict__ deg,
                               const float* __restrict__ xw, float* __restrict__ agg, int C)
{
    int t = blockIdx.x * 256 + threadIdx.x;
    if (t >= (EG + NBATCH) * C) return;
    int e = t / C, c = t - e * C;
    int r  = (e < EG) ? ei[e]       : (e - EG);
    int cc = (e < EG) ? ei[EG + e]  : (e - EG);
    float nrm = rsqrtf(deg[r]) * rsqrtf(deg[cc]);
    atomicAdd(&agg[(size_t)cc * C + c], xw[(size_t)r * C + c] * nrm);
}
__global__ void bias_bn_gelu_kernel(const float* __restrict__ agg, const float* __restrict__ gb,
                                    const float* __restrict__ bng, const float* __restrict__ bnb,
                                    float* __restrict__ xo, int C)
{
    int t = blockIdx.x * 256 + threadIdx.x;
    if (t >= NBATCH * C) return;
    int c = t % C;
    float v = agg[t] + gb[c];
    v = v * (bng[c] * rsqrtf(1.0f + 1e-5f)) + bnb[c];
    xo[t] = gelu_tanh(v);
}
__global__ void edge_cls_kernel(const float* __restrict__ x2, const int* __restrict__ edges,
                                const int* __restrict__ yl, const float* __restrict__ Wc,
                                const float* __restrict__ bc, float* __restrict__ out)
{
    int e = blockIdx.x * 256 + threadIdx.x;
    if (e >= EC) return;
    int a = edges[2 * e], b = edges[2 * e + 1];
    float l0 = bc[0], l1 = bc[1];
    for (int j = 0; j < G2DIM; j++) {
        float xa = x2[(size_t)a * G2DIM + j];
        l0 += xa * Wc[j * 2 + 0];
        l1 += xa * Wc[j * 2 + 1];
    }
    for (int j = 0; j < G2DIM; j++) {
        float xb = x2[(size_t)b * G2DIM + j];
        l0 += xb * Wc[(G2DIM + j) * 2 + 0];
        l1 += xb * Wc[(G2DIM + j) * 2 + 1];
    }
    int pred = (l1 > l0) ? 1 : 0;
    float mx = fmaxf(l0, l1);
    float lse = mx + logf(expf(l0 - mx) + expf(l1 - mx));
    int y = yl[e];
    float logp = (y ? l1 : l0) - lse;
    atomicAdd(&out[0], -logp * (1.0f / EC));
    out[1 + e] = (float)pred;
    out[1 + EC + e] = (float)y;
}

// ---------------------------------------------------------------------------
// Orchestration
// ---------------------------------------------------------------------------
extern "C" void kernel_launch(void* const* d_in, const int* in_sizes, int n_in,
                              void* d_out, int out_size, void* d_ws, size_t ws_size,
                              hipStream_t stream)
{
    (void)in_sizes; (void)n_in; (void)out_size; (void)ws_size;
    auto P = [&](int i) { return (const float*)d_in[i]; };

    const int* ids    = (const int*)d_in[0];
    const int* mask   = (const int*)d_in[1];
    const int* seg    = (const int*)d_in[2];
    const int* ei     = (const int*)d_in[3];
    const int* aedges = (const int*)d_in[4];
    const int* ylab   = (const int*)d_in[5];
    const float* word = P(6);
    const float* pos  = P(7);
    const float* typ  = P(8);
    const float* emb_g = P(9);
    const float* emb_b = P(10);
    // layers: base 11, 16 tensors each:
    //  0 Wq 1 bq 2 Wk 3 bk 4 Wv 5 bv 6 Wo 7 bo 8 ln1g 9 ln1b 10 Wi 11 bi 12 Wf 13 bf 14 ln2g 15 ln2b
    const float* Wp = P(43); const float* bp = P(44);
    const float* g1W = P(45); const float* g1b = P(46);
    const float* bn1g = P(47); const float* bn1b = P(48);
    const float* g2W = P(49); const float* g2b = P(50);
    const float* bn2g = P(51); const float* bn2b = P(52);
    const float* clsW = P(53); const float* clsb_p = P(54);

    // ---- workspace carve ----
    size_t off = 0;
    auto alloc = [&](size_t bytes) -> char* {
        char* p = (char*)d_ws + off;
        off += (bytes + 255) & ~(size_t)255;
        return p;
    };
    u16* hbuf = (u16*)alloc((size_t)MTOK * HID * 2);
    u16* qbuf = (u16*)alloc((size_t)MTOK * HID * 2);
    u16* kbuf = (u16*)alloc((size_t)MTOK * HID * 2);
    u16* vbuf = (u16*)alloc((size_t)MTOK * HID * 2);
    u16* cbuf = (u16*)alloc((size_t)MTOK * HID * 2);
    u16* t1   = (u16*)alloc((size_t)MTOK * HID * 2);
    u16* ffb  = (u16*)alloc((size_t)MTOK * FFD * 2);
    u16* wqt[2], *wkt[2], *wvt[2], *wot[2], *wit[2], *wft[2];
    for (int l = 0; l < 2; l++) {
        wqt[l] = (u16*)alloc((size_t)HID * HID * 2);
        wkt[l] = (u16*)alloc((size_t)HID * HID * 2);
        wvt[l] = (u16*)alloc((size_t)HID * HID * 2);
        wot[l] = (u16*)alloc((size_t)HID * HID * 2);
        wit[l] = (u16*)alloc((size_t)HID * FFD * 2);
        wft[l] = (u16*)alloc((size_t)HID * FFD * 2);
    }
    u16* wpt  = (u16*)alloc((size_t)HID * HID * 2);
    u16* clsb = (u16*)alloc((size_t)NBATCH * HID * 2);
    float* xw1  = (float*)alloc((size_t)NBATCH * G1DIM * 4);
    float* agg1 = (float*)alloc((size_t)NBATCH * G1DIM * 4);
    float* x1   = (float*)alloc((size_t)NBATCH * G1DIM * 4);
    float* xw2  = (float*)alloc((size_t)NBATCH * G2DIM * 4);
    float* agg2 = (float*)alloc((size_t)NBATCH * G2DIM * 4);
    float* x2   = (float*)alloc((size_t)NBATCH * G2DIM * 4);
    float* deg  = (float*)alloc((size_t)NBATCH * 4);

    auto tgrid = [](size_t n) { return (unsigned)((n + 255) / 256); };

    // ---- weight convert + transpose to bf16 [N,K] ----
    for (int l = 0; l < 2; l++) {
        int base = 11 + 16 * l;
        cvt_transpose_kernel<<<tgrid((size_t)HID * HID), 256, 0, stream>>>(P(base + 0),  wqt[l], HID, HID);
        cvt_transpose_kernel<<<tgrid((size_t)HID * HID), 256, 0, stream>>>(P(base + 2),  wkt[l], HID, HID);
        cvt_transpose_kernel<<<tgrid((size_t)HID * HID), 256, 0, stream>>>(P(base + 4),  wvt[l], HID, HID);
        cvt_transpose_kernel<<<tgrid((size_t)HID * HID), 256, 0, stream>>>(P(base + 6),  wot[l], HID, HID);
        cvt_transpose_kernel<<<tgrid((size_t)HID * FFD), 256, 0, stream>>>(P(base + 10), wit[l], HID, FFD);
        cvt_transpose_kernel<<<tgrid((size_t)HID * FFD), 256, 0, stream>>>(P(base + 12), wft[l], FFD, HID);
    }
    cvt_transpose_kernel<<<tgrid((size_t)HID * HID), 256, 0, stream>>>(Wp, wpt, HID, HID);

    // ---- embeddings + LN ----
    embed_ln_kernel<<<MTOK, 256, 0, stream>>>(ids, seg, word, pos, typ, emb_g, emb_b, hbuf);

    // ---- BERT layers ----
    dim3 gHH(MTOK / 128, HID / 128);    // 1024 x 6
    dim3 gHF(MTOK / 128, FFD / 128);    // 1024 x 24
    for (int l = 0; l < 2; l++) {
        int base = 11 + 16 * l;
        const float* bq = P(base + 1); const float* bk = P(base + 3);
        const float* bv = P(base + 5); const float* bo = P(base + 7);
        const float* ln1g = P(base + 8);  const float* ln1b = P(base + 9);
        const float* bi = P(base + 11); const float* bf = P(base + 13);
        const float* ln2g = P(base + 14); const float* ln2b = P(base + 15);

        gemm_bf16_wmma<<<gHH, 256, 0, stream>>>(hbuf, HID, wqt[l], HID, bq, nullptr, 0, qbuf, HID, HID, 0);
        gemm_bf16_wmma<<<gHH, 256, 0, stream>>>(hbuf, HID, wkt[l], HID, bk, nullptr, 0, kbuf, HID, HID, 0);
        gemm_bf16_wmma<<<gHH, 256, 0, stream>>>(hbuf, HID, wvt[l], HID, bv, nullptr, 0, vbuf, HID, HID, 0);

        attention_kernel<<<NBATCH * NHEADS, 128, 0, stream>>>(qbuf, kbuf, vbuf, mask, cbuf);

        gemm_bf16_wmma<<<gHH, 256, 0, stream>>>(cbuf, HID, wot[l], HID, bo, hbuf, HID, t1, HID, HID, 0);
        ln_kernel<<<MTOK, 256, 0, stream>>>(t1, ln1g, ln1b, hbuf);

        gemm_bf16_wmma<<<gHF, 256, 0, stream>>>(hbuf, HID, wit[l], HID, bi, nullptr, 0, ffb, FFD, HID, 1);
        gemm_bf16_wmma<<<gHH, 256, 0, stream>>>(ffb, FFD, wft[l], FFD, bf, hbuf, HID, t1, HID, FFD, 0);
        ln_kernel<<<MTOK, 256, 0, stream>>>(t1, ln2g, ln2b, hbuf);
    }

    // ---- pooler: tanh(CLS @ Wp + bp); CLS rows have stride SEQ*HID in hbuf ----
    dim3 gPool(NBATCH / 128, HID / 128);
    gemm_bf16_wmma<<<gPool, 256, 0, stream>>>(hbuf, SEQ * HID, wpt, HID, bp, nullptr, 0,
                                              clsb, HID, HID, 2);

    // ---- GCN layer 1 ----
    gemm_bf16f32_plain<<<tgrid((size_t)NBATCH * G1DIM), 256, 0, stream>>>(clsb, g1W, xw1, NBATCH, HID, G1DIM);
    hipMemsetAsync(deg,  0, (size_t)NBATCH * 4, stream);
    hipMemsetAsync(agg1, 0, (size_t)NBATCH * G1DIM * 4, stream);
    hipMemsetAsync(agg2, 0, (size_t)NBATCH * G2DIM * 4, stream);
    deg_kernel<<<tgrid(EG + NBATCH), 256, 0, stream>>>(ei, deg);
    scatter_kernel<<<tgrid((size_t)(EG + NBATCH) * G1DIM), 256, 0, stream>>>(ei, deg, xw1, agg1, G1DIM);
    bias_bn_gelu_kernel<<<tgrid((size_t)NBATCH * G1DIM), 256, 0, stream>>>(agg1, g1b, bn1g, bn1b, x1, G1DIM);

    // ---- GCN layer 2 ----
    gemm_f32_plain<<<tgrid((size_t)NBATCH * G2DIM), 256, 0, stream>>>(x1, g2W, xw2, NBATCH, G1DIM, G2DIM);
    scatter_kernel<<<tgrid((size_t)(EG + NBATCH) * G2DIM), 256, 0, stream>>>(ei, deg, xw2, agg2, G2DIM);
    bias_bn_gelu_kernel<<<tgrid((size_t)NBATCH * G2DIM), 256, 0, stream>>>(agg2, g2b, bn2g, bn2b, x2, G2DIM);

    // ---- edge classifier + loss ----
    hipMemsetAsync(d_out, 0, 4, stream);
    edge_cls_kernel<<<tgrid(EC), 256, 0, stream>>>(x2, aedges, ylab, clsW, clsb_p, (float*)d_out);
}
